// GAT_52381421142172
// MI455X (gfx1250) — matrix operand order repaired and checked by reading, not compiled, not run on hardware
//
#include <hip/hip_runtime.h>
#include <hip/hip_bf16.h>

#define NN 50000
#define NE 800000
#define NEG_SLOPE 0.2f
#define IN_DIM 128   // all three layers consume 128-wide inputs

typedef __attribute__((ext_vector_type(16))) __bf16 v16bf;
typedef __attribute__((ext_vector_type(8)))  __bf16 v8bf;
typedef __attribute__((ext_vector_type(8)))  float  v8f;

__device__ __forceinline__ unsigned short f2bf_bits(float f) {
    unsigned u = __float_as_uint(f);
    unsigned r = (u + 0x7FFFu + ((u >> 16) & 1u)) >> 16;   // round-to-nearest-even
    return (unsigned short)r;
}

// ---------------------------------------------------------------- fill
__global__ void gat_fill(float* __restrict__ p, float v, int n) {
    int i = blockIdx.x * blockDim.x + threadIdx.x;
    if (i < n) p[i] = v;
}

// ---------------------------------------------------------------- f32 -> bf16 (row-major copy)
__global__ void gat_cvt_bf16(const float* __restrict__ in,
                             unsigned short* __restrict__ out, int n) {
    int i = blockIdx.x * blockDim.x + threadIdx.x;
    if (i < n) out[i] = f2bf_bits(in[i]);
}

// ---------------------------------------------------------------- W [in,out] f32 -> W^T [out,in] bf16
__global__ void gat_cvt_wt(const float* __restrict__ W,
                           unsigned short* __restrict__ wt,
                           int in_dim, int out_dim) {
    int i = blockIdx.x * blockDim.x + threadIdx.x;
    if (i >= in_dim * out_dim) return;
    int n = i / in_dim, k = i - n * in_dim;
    wt[i] = f2bf_bits(W[(size_t)k * out_dim + n]);
}

// ---------------------------------------------------------------- GEMM: feat = x @ W
// xb:  [NN, IN_DIM] bf16 row-major;  wtb: [out_dim, IN_DIM] bf16 (W transposed).
// One wave per 16x64 output strip: A fragment loaded once per K-chunk (2x b128),
// reused across 4 N-tiles (each 2x b128 for B) -> 16 v_wmma per wave, fully unrolled.
// Fragment layouts per CDNA5 ISA:
//   A 16x32: lanes 0-15 hold K=0..7 (VGPR0-3) and K=16..23 (VGPR4-7); lanes 16-31: +8.
//   B 32x16 as loaded here (A likewise, not the doc order): lanes 0-15 hold K=0..15; lanes 16-31 K=16..31 (column = lane&15).
__global__ void gat_gemm_wmma(const unsigned short* __restrict__ xb,
                              const unsigned short* __restrict__ wtb,
                              float* __restrict__ feat,
                              int out_dim) {
    const int lane = threadIdx.x;
    const int tm = blockIdx.x;
    const int tn4 = blockIdx.y;              // strip of 4 N-tiles
    const int hi = lane >> 4;                // 0: lanes 0-15, 1: lanes 16-31
    const int lo = lane & 15;
    const int arow = tm * 16 + lo;
    const int aoff = hi * 16;                // A K-offset (contiguous, matches B)
    const int bklo = hi * 16;                // B K-offset

    const __bf16* xrow = (const __bf16*)xb + (size_t)arow * IN_DIM;
    const __bf16* wbase = (const __bf16*)wtb;

    v8f acc0 = {}, acc1 = {}, acc2 = {}, acc3 = {};
#pragma unroll
    for (int kc = 0; kc < IN_DIM / 32; ++kc) {
        const int kk = kc * 32;
        v8bf a0 = *(const v8bf*)(xrow + kk + aoff);
        v8bf a1 = *(const v8bf*)(xrow + kk + aoff + 8);
        v16bf a = __builtin_shufflevector(a0, a1,
                      0,1,2,3,4,5,6,7,8,9,10,11,12,13,14,15);
#pragma unroll
        for (int t = 0; t < 4; ++t) {
            const int bcol = (tn4 * 4 + t) * 16 + lo;
            const __bf16* wrow = wbase + (size_t)bcol * IN_DIM + kk + bklo;
            v8bf b0 = *(const v8bf*)(wrow);
            v8bf b1 = *(const v8bf*)(wrow + 8);
            v16bf b = __builtin_shufflevector(b0, b1,
                          0,1,2,3,4,5,6,7,8,9,10,11,12,13,14,15);
            v8f c = (t == 0) ? acc0 : (t == 1) ? acc1 : (t == 2) ? acc2 : acc3;
            c = __builtin_amdgcn_wmma_f32_16x16x32_bf16(
                    false, a, false, b, (short)0, c, false, false);
            if      (t == 0) acc0 = c;
            else if (t == 1) acc1 = c;
            else if (t == 2) acc2 = c;
            else             acc3 = c;
        }
    }
    // C/D layout: VGPR r -> row (r + 8*hi), column lo
    const int rbase = tm * 16 + hi * 8;
#pragma unroll
    for (int t = 0; t < 4; ++t) {
        v8f c = (t == 0) ? acc0 : (t == 1) ? acc1 : (t == 2) ? acc2 : acc3;
        const int ccol = (tn4 * 4 + t) * 16 + lo;
#pragma unroll
        for (int r = 0; r < 8; ++r)
            feat[(size_t)(rbase + r) * out_dim + ccol] = c[r];
    }
}

// ---------------------------------------------------------------- el/er = feat . al/ar
__global__ void gat_attn(const float* __restrict__ feat,
                         const float* __restrict__ al,
                         const float* __restrict__ ar,
                         float* __restrict__ el, float* __restrict__ er,
                         int H, int D) {
    int i = blockIdx.x * blockDim.x + threadIdx.x;
    if (i >= NN * H) return;
    int n = i / H, h = i - n * H;
    const float* f = feat + (size_t)n * H * D + h * D;
    const float* a = al + h * D;
    const float* c = ar + h * D;
    float sl = 0.f, sr = 0.f;
    for (int d = 0; d < D; ++d) { float v = f[d]; sl += v * a[d]; sr += v * c[d]; }
    el[i] = sl; er[i] = sr;
}

// float atomic-max via monotone int/uint encoding (safe under interleaving)
__device__ __forceinline__ void atomic_max_f32(float* addr, float val) {
    if (val >= 0.f) atomicMax((int*)addr, __float_as_int(val));
    else            atomicMin((unsigned int*)addr, __float_as_uint(val));
}

// ---------------------------------------------------------------- edge pass 1: scores + segment max
__global__ void gat_edge_score(const float* __restrict__ el,
                               const float* __restrict__ er,
                               const int* __restrict__ src,
                               const int* __restrict__ dst,
                               float* __restrict__ esc,
                               float* __restrict__ emax, int H) {
    int e = blockIdx.x * blockDim.x + threadIdx.x;
    if (e >= NE) return;
    int s = src[e], d = dst[e];
    for (int h = 0; h < H; ++h) {
        float v = el[s * H + h] + er[d * H + h];
        v = v > 0.f ? v : NEG_SLOPE * v;                 // leaky relu
        esc[(size_t)e * H + h] = v;
        atomic_max_f32(&emax[d * H + h], v);
    }
}

// ---------------------------------------------------------------- edge pass 2: exp + segment sum
__global__ void gat_edge_weight(const int* __restrict__ dst,
                                float* __restrict__ esc,
                                const float* __restrict__ emax,
                                float* __restrict__ denom, int H) {
    int e = blockIdx.x * blockDim.x + threadIdx.x;
    if (e >= NE) return;
    int d = dst[e];
    for (int h = 0; h < H; ++h) {
        float w = __expf(esc[(size_t)e * H + h] - emax[d * H + h]);
        esc[(size_t)e * H + h] = w;
        atomicAdd(&denom[d * H + h], w);
    }
}

// ---------------------------------------------------------------- edge pass 3: message scatter
// one block per edge, blockDim = H*D threads (one per feature)
__global__ void gat_edge_aggregate(const float* __restrict__ feat,
                                   const float* __restrict__ esc,
                                   const float* __restrict__ denom,
                                   const int* __restrict__ src,
                                   const int* __restrict__ dst,
                                   float* __restrict__ out, int H, int D) {
    int e = blockIdx.x;
    int t = threadIdx.x;                // feature index 0..H*D-1
    int h = t / D;
    int s = src[e], d = dst[e];
    float dn = denom[d * H + h];
    float alpha = esc[(size_t)e * H + h] / fmaxf(dn, 1e-12f);
    atomicAdd(&out[(size_t)d * H * D + t], feat[(size_t)s * H * D + t] * alpha);
}

// ---------------------------------------------------------------- bias (+relu)
__global__ void gat_bias_act(float* __restrict__ out, const float* __restrict__ b,
                             int F, int act, int n) {
    int i = blockIdx.x * blockDim.x + threadIdx.x;
    if (i >= n) return;
    float v = out[i] + b[i % F];
    out[i] = act ? fmaxf(v, 0.f) : v;
}

// ---------------------------------------------------------------- host-side layer driver
static void run_layer(const float* x, const int* src, const int* dst,
                      const float* W, const float* al, const float* ar, const float* b,
                      float* out, float* feat, float* el, float* er,
                      float* emax, float* denom, float* esc,
                      unsigned short* xb, unsigned short* wtb,
                      int H, int D, int act, hipStream_t stream) {
    const int F = H * D;
    const float ninf = -__builtin_inff();
    gat_fill<<<(NN * F + 255) / 256, 256, 0, stream>>>(out, 0.f, NN * F);
    gat_fill<<<(NN * H + 255) / 256, 256, 0, stream>>>(emax, ninf, NN * H);
    gat_fill<<<(NN * H + 255) / 256, 256, 0, stream>>>(denom, 0.f, NN * H);

    gat_cvt_bf16<<<(NN * IN_DIM + 255) / 256, 256, 0, stream>>>(x, xb, NN * IN_DIM);
    gat_cvt_wt<<<(IN_DIM * F + 255) / 256, 256, 0, stream>>>(W, wtb, IN_DIM, F);

    dim3 ggrid(NN / 16, F / 64);
    gat_gemm_wmma<<<ggrid, 32, 0, stream>>>(xb, wtb, feat, F);

    gat_attn<<<(NN * H + 255) / 256, 256, 0, stream>>>(feat, al, ar, el, er, H, D);
    gat_edge_score<<<(NE + 255) / 256, 256, 0, stream>>>(el, er, src, dst, esc, emax, H);
    gat_edge_weight<<<(NE + 255) / 256, 256, 0, stream>>>(dst, esc, emax, denom, H);
    gat_edge_aggregate<<<NE, F, 0, stream>>>(feat, esc, denom, src, dst, out, H, D);
    gat_bias_act<<<(NN * F + 255) / 256, 256, 0, stream>>>(out, b, F, act, NN * F);
}

extern "C" void kernel_launch(void* const* d_in, const int* in_sizes, int n_in,
                              void* d_out, int out_size, void* d_ws, size_t ws_size,
                              hipStream_t stream) {
    const float* features = (const float*)d_in[0];
    const int*   src      = (const int*)d_in[1];
    const int*   dst      = (const int*)d_in[2];
    const float* W0 = (const float*)d_in[3];
    const float* al0 = (const float*)d_in[4];
    const float* ar0 = (const float*)d_in[5];
    const float* b0 = (const float*)d_in[6];
    const float* W1 = (const float*)d_in[7];
    const float* al1 = (const float*)d_in[8];
    const float* ar1 = (const float*)d_in[9];
    const float* b1 = (const float*)d_in[10];
    const float* W2 = (const float*)d_in[11];
    const float* al2 = (const float*)d_in[12];
    const float* ar2 = (const float*)d_in[13];
    const float* b2 = (const float*)d_in[14];

    float* w = (float*)d_ws;
    float* bufA  = w; w += (size_t)NN * 128;
    float* bufB  = w; w += (size_t)NN * 128;
    float* feat  = w; w += (size_t)NN * 128;
    float* el    = w; w += (size_t)NN * 4;
    float* er    = w; w += (size_t)NN * 4;
    float* emax  = w; w += (size_t)NN * 4;
    float* denom = w; w += (size_t)NN * 4;
    float* esc   = w; w += (size_t)NE * 4;
    unsigned short* xb  = (unsigned short*)w;          // NN*128 bf16
    unsigned short* wtb = xb + (size_t)NN * 128;       // 128*128 bf16 (max)

    // layer 0: [NN,128] -> [NN,128], H=4, D=32, relu
    run_layer(features, src, dst, W0, al0, ar0, b0,
              bufA, feat, el, er, emax, denom, esc, xb, wtb,
              4, 32, 1, stream);
    // layer 1: [NN,128] -> [NN,128], H=4, D=32, relu
    run_layer(bufA, src, dst, W1, al1, ar1, b1,
              bufB, feat, el, er, emax, denom, esc, xb, wtb,
              4, 32, 1, stream);
    // layer 2: [NN,128] -> [NN,64], H=1, D=64, no act; accumulate into d_out
    run_layer(bufB, src, dst, W2, al2, ar2, b2,
              (float*)d_out, feat, el, er, emax, denom, esc, xb, wtb,
              1, 64, 0, stream);
}
